// FaceAttention2_16217796509901
// MI455X (gfx1250) — matrix operand order repaired and checked by reading, not compiled, not run on hardware
//
#include <hip/hip_runtime.h>
#include <math.h>

typedef __attribute__((ext_vector_type(16))) __bf16          v16bf;
typedef __attribute__((ext_vector_type(16))) unsigned short  v16u;
typedef __attribute__((ext_vector_type(8)))  unsigned short  v8u;
typedef __attribute__((ext_vector_type(8)))  float           v8f;

#define B_  32
#define S_  2048
#define D_  512
#define K_  2560           // DIM1 + DIM2
#define N_  512            // DIM2
#define KTILES (K_ / 32)   // 80 WMMA K-steps total
#define NTILES (N_ / 16)   // 32 N-tiles
#define KCHUNK 640         // K staged per LDS round
#define KCHUNK_STEPS (KCHUNK / 32)  // 20
#define NCHUNKS (K_ / KCHUNK)       // 4

// f32 -> bf16 (round-to-nearest-even), as a raw bit pattern.
static __device__ __forceinline__ unsigned short f32_to_bf16(float f) {
    unsigned int u = __float_as_uint(f);
    u += 0x7FFFu + ((u >> 16) & 1u);
    return (unsigned short)(u >> 16);
}

// ---------------------------------------------------------------------------
// Kernel 0: convert W1 (f32 [512 x 2560], row-major [n][k]) into bf16 bits,
// pre-swizzled so each wave32 lane's WMMA B-fragment (32x16 K-major tile)
// is 16 contiguous bf16 (32 bytes).
// Layout: Bws[((ntile*KTILES + kt)*512) + lane*16 + j]
//   n = ntile*16 + (lane & 15)
//   k = kt*32 + ((lane >> 4) << 4) + j     (lanes 16-31 hold K=16..31)
// ---------------------------------------------------------------------------
__global__ __launch_bounds__(256) void prep_w1(const float* __restrict__ W1,
                                               unsigned short* __restrict__ Bws) {
    int t    = blockIdx.x * 256 + threadIdx.x;
    int j    = t & 15;
    int lane = (t >> 4) & 31;
    int blk  = t >> 9;
    int kt   = blk % KTILES;
    int ntile = blk / KTILES;
    if (ntile >= NTILES) return;
    int n = ntile * 16 + (lane & 15);
    int k = kt * 32 + ((lane >> 4) << 4) + j;
    Bws[t] = f32_to_bf16(W1[(size_t)n * K_ + k]);
}

// ---------------------------------------------------------------------------
// Kernel 1: h = tanh(concat(face,fc,img,lab,mem) @ W1^T + b1); e = h . W2
// One workgroup (8 waves) per 16-row M-tile. A staged in LDS as bf16 bits in
// 640-wide K chunks; each wave computes 4 N-tiles via v_wmma_f32_16x16x32_bf16.
// ---------------------------------------------------------------------------
__global__ __launch_bounds__(256) void gemm_tanh_dot(
    const float* __restrict__ face, const float* __restrict__ fc,
    const float* __restrict__ img,  const float* __restrict__ lab,
    const float* __restrict__ mem,  const unsigned short* __restrict__ Bws,
    const float* __restrict__ b1,   const float* __restrict__ W2,
    float* __restrict__ e_out) {
    __shared__ __align__(16) unsigned short At[16 * KCHUNK];   // 20 KB
    __shared__ float e_lds[16];

    const int tid   = threadIdx.x;
    const int mtile = blockIdx.x;              // 0..4095
    const int b     = (mtile * 16) / S_;
    const int sbase = (mtile * 16) % S_;       // S_ % 16 == 0, tile never crosses batch

    if (tid < 16) e_lds[tid] = 0.0f;

    const int lane = tid & 31;
    const int wave = tid >> 5;
    const int r    = lane & 15;

    v8f acc[4];
#pragma unroll
    for (int nt = 0; nt < 4; ++nt)
#pragma unroll
        for (int i = 0; i < 8; ++i) acc[nt][i] = 0.0f;

    for (int kc = 0; kc < NCHUNKS; ++kc) {
        __syncthreads();
        // ---- stage A chunk (f32 -> bf16 bits, 5-way concat gather) ----
        for (int idx = tid; idx < 16 * KCHUNK; idx += 256) {
            int rr = idx / KCHUNK;
            int kk = idx - rr * KCHUNK;
            int k  = kc * KCHUNK + kk;
            float v;
            if (k < 3 * D_) {
                const float* src = (k < D_) ? face : (k < 2 * D_) ? fc : img;
                v = src[((size_t)(b * S_ + sbase + rr)) * D_ + (k & (D_ - 1))];
            } else if (k < 4 * D_) {
                v = lab[b * D_ + (k - 3 * D_)];
            } else {
                v = mem[b * D_ + (k - 4 * D_)];
            }
            At[idx] = f32_to_bf16(v);
        }
        __syncthreads();

        const int klo_base = (lane >> 4) << 4;    // lanes 16-31: +16
#pragma unroll
        for (int nt = 0; nt < 4; ++nt) {
            const int ntile = wave * 4 + nt;
            const unsigned short* bbase =
                Bws + (size_t)(ntile * KTILES + kc * KCHUNK_STEPS) * 512 + lane * 16;
            for (int ks = 0; ks < KCHUNK_STEPS; ++ks) {
                // A fragment: lane holds row r, K = klo..klo+15
                int klo = ks * 32 + klo_base;
                v8u alo = *(const v8u*)&At[r * KCHUNK + klo];
                v8u ahi = *(const v8u*)&At[r * KCHUNK + klo + 8];
                v16u ar;
#pragma unroll
                for (int i = 0; i < 8; ++i) { ar[i] = alo[i]; ar[i + 8] = ahi[i]; }
                // B fragment: one contiguous 32-byte read (pre-swizzled)
                v16u br = *(const v16u*)(bbase + (size_t)ks * 512);
                v16bf a  = __builtin_bit_cast(v16bf, ar);
                v16bf bf = __builtin_bit_cast(v16bf, br);
                acc[nt] = __builtin_amdgcn_wmma_f32_16x16x32_bf16(
                    false, a, false, bf, (short)0, acc[nt], false, false);
            }
        }
    }

    // ---- e[m] = sum_n tanh(h[m,n] + b1[n]) * W2[n] ----
    // C layout: lanes 0-15 -> (M=v, N=lane); lanes 16-31 -> (M=v+8, N=lane-16)
#pragma unroll
    for (int nt = 0; nt < 4; ++nt) {
        const int n  = (wave * 4 + nt) * 16 + r;
        const float w2n = W2[n];
        const float b1n = b1[n];
#pragma unroll
        for (int v = 0; v < 8; ++v) {
            float val = tanhf(acc[nt][v] + b1n) * w2n;
            // reduce over the 16 columns held by this half-wave
            val += __shfl_xor(val, 1);
            val += __shfl_xor(val, 2);
            val += __shfl_xor(val, 4);
            val += __shfl_xor(val, 8);
            if (r == 0) {
                int m = v + ((lane >> 4) << 3);
                atomicAdd(&e_lds[m], val);
            }
        }
    }
    __syncthreads();
    if (tid < 16) e_out[(size_t)mtile * 16 + tid] = e_lds[tid];
}

// ---------------------------------------------------------------------------
// Kernel 2: per-batch softmax over S, mask, renormalize (+EPS), write alpha,
// then context[b,d] = sum_s alpha[b,s] * face[b,s,d].
// ---------------------------------------------------------------------------
__global__ __launch_bounds__(256) void softmax_context(
    const float* __restrict__ e_in, const int* __restrict__ mask,
    const float* __restrict__ face, float* __restrict__ alpha_out,
    float* __restrict__ ctx_out) {
    __shared__ float red[256];
    __shared__ float al[S_];
    const int b   = blockIdx.x;
    const int tid = threadIdx.x;
    const float* e = e_in + (size_t)b * S_;

    // max
    float m = -INFINITY;
    for (int i = tid; i < S_; i += 256) m = fmaxf(m, e[i]);
    red[tid] = m; __syncthreads();
    for (int st = 128; st > 0; st >>= 1) {
        if (tid < st) red[tid] = fmaxf(red[tid], red[tid + st]);
        __syncthreads();
    }
    const float mmax = red[0]; __syncthreads();

    // exp + sum
    float s = 0.0f;
    for (int i = tid; i < S_; i += 256) {
        float ex = expf(e[i] - mmax);
        al[i] = ex; s += ex;
    }
    red[tid] = s; __syncthreads();
    for (int st = 128; st > 0; st >>= 1) {
        if (tid < st) red[tid] += red[tid + st];
        __syncthreads();
    }
    const float inv = 1.0f / red[0]; __syncthreads();

    // mask + masked sum
    float ms = 0.0f;
    for (int i = tid; i < S_; i += 256) {
        float a = al[i] * inv * (float)mask[(size_t)b * S_ + i];
        al[i] = a; ms += a;
    }
    red[tid] = ms; __syncthreads();
    for (int st = 128; st > 0; st >>= 1) {
        if (tid < st) red[tid] += red[tid + st];
        __syncthreads();
    }
    const float inv2 = 1.0f / (red[0] + 1e-5f); __syncthreads();

    for (int i = tid; i < S_; i += 256) {
        float a = al[i] * inv2;
        al[i] = a;
        alpha_out[(size_t)b * S_ + i] = a;
    }
    __syncthreads();

    // context
    for (int d = tid; d < D_; d += 256) {
        const float* fb = face + (size_t)b * S_ * D_ + d;
        float c0 = 0.f, c1 = 0.f, c2 = 0.f, c3 = 0.f;
        for (int ss = 0; ss < S_; ss += 4) {
            c0 += al[ss + 0] * fb[(size_t)(ss + 0) * D_];
            c1 += al[ss + 1] * fb[(size_t)(ss + 1) * D_];
            c2 += al[ss + 2] * fb[(size_t)(ss + 2) * D_];
            c3 += al[ss + 3] * fb[(size_t)(ss + 3) * D_];
        }
        ctx_out[(size_t)b * D_ + d] = (c0 + c1) + (c2 + c3);
    }
}

// ---------------------------------------------------------------------------
extern "C" void kernel_launch(void* const* d_in, const int* in_sizes, int n_in,
                              void* d_out, int out_size, void* d_ws, size_t ws_size,
                              hipStream_t stream) {
    const float* fc   = (const float*)d_in[0];
    const float* img  = (const float*)d_in[1];
    const float* lab  = (const float*)d_in[2];
    const float* mem  = (const float*)d_in[3];
    const float* face = (const float*)d_in[4];
    const int*   mask = (const int*)d_in[5];
    const float* W1   = (const float*)d_in[6];
    const float* b1   = (const float*)d_in[7];
    const float* W2   = (const float*)d_in[8];

    unsigned short* Bws = (unsigned short*)d_ws;                  // 2.62 MB
    float* e = (float*)((char*)d_ws + (size_t)N_ * K_ * 2);       // 256 KB

    float* alpha_out = (float*)d_out;                 // [B,S]
    float* ctx_out   = alpha_out + (size_t)B_ * S_;   // [B,D]

    prep_w1<<<(N_ * K_) / 256, 256, 0, stream>>>(W1, Bws);
    gemm_tanh_dot<<<(B_ * S_) / 16, 256, 0, stream>>>(face, fc, img, lab, mem,
                                                      Bws, b1, W2, e);
    softmax_context<<<B_, 256, 0, stream>>>(e, mask, face, alpha_out, ctx_out);
}
